// Encoder_75204877353660
// MI455X (gfx1250) — matrix operand order repaired and checked by reading, not compiled, not run on hardware
//
#include <hip/hip_runtime.h>
#include <hip/hip_bf16.h>

typedef __attribute__((ext_vector_type(16))) _Float16 v16h;
typedef __attribute__((ext_vector_type(8)))  _Float16 v8h;
typedef __attribute__((ext_vector_type(8)))  float    v8f;

#define BB    32
#define EE    256
#define NN    512          // 2E nodes
#define SS    255          // E-1 steps
#define HID   64           // per-head features
#define HOUT  256          // H*HID
#define NHEAD 4

// -------------------------------------------------------------------------
// adjacency: adj[b,r,i,j] closed form. base13 = complete minus eye, static
// pairs (|i-j|==2) carry rel 11/12, dynamic pairs (diff 3, lo%4==0) or
// (diff 1, lo%4==3) carry rel shot_type[b,s]; both zero rel 13.
// -------------------------------------------------------------------------
__global__ void adj_kernel(const int* __restrict__ shot, float* __restrict__ adj) {
  size_t flat = (size_t)blockIdx.x * blockDim.x + threadIdx.x;
  size_t total = (size_t)BB * 14 * NN * (NN / 4);
  if (flat >= total) return;
  int jq = (int)(flat & 127);          // 128 float4 per row
  size_t t = flat >> 7;
  int i = (int)(t & 511); t >>= 9;
  int r = (int)(t % 14);
  int b = (int)(t / 14);
  float4 v;
  float* vv = &v.x;
  #pragma unroll
  for (int q = 0; q < 4; ++q) {
    int j = jq * 4 + q;
    int lo = min(i, j), hi = max(i, j), d = hi - lo;
    bool stat = (d == 2);
    bool dyn = false; int s = -1;
    if (d == 3 && (lo & 3) == 0)      { s = lo >> 1;       dyn = true; }
    else if (d == 1 && (lo & 3) == 3) { s = (lo - 1) >> 1; dyn = true; }
    if (dyn && s >= SS) dyn = false;
    float val = 0.f;
    if (r == 13) {
      val = (i != j && !stat && !dyn) ? 1.f : 0.f;
    } else if (r >= 11) {
      if (stat) {
        int rel = 11 + ((lo ^ (lo >> 1)) & 1);
        val = (rel == r) ? 1.f : 0.f;
      }
    } else {
      if (dyn && shot[b * SS + s] == r) val = 1.f;
    }
    vv[q] = val;
  }
  *(float4*)(adj + flat * 4) = v;
}

// -------------------------------------------------------------------------
// weight pre-transpose + f32->f16: Wt[c][k] = (f16) W[k][c], c in [0,256)
// -------------------------------------------------------------------------
__global__ void wtrans_kernel(const float* __restrict__ W, _Float16* __restrict__ Wt, int K) {
  int idx = blockIdx.x * blockDim.x + threadIdx.x;     // c*K + k
  if (idx >= 256 * K) return;
  int c = idx / K, k = idx - c * K;
  Wt[idx] = (_Float16)W[k * 256 + c];
}

// -------------------------------------------------------------------------
// node embedding: x[b,n,:] = relu(coord@Wc+bc) || emb[player[b,n&1]] -> @Wm+bm
// -------------------------------------------------------------------------
__global__ void embed_kernel(const int* __restrict__ player,
                             const float* __restrict__ Ax, const float* __restrict__ Ay,
                             const float* __restrict__ Bx, const float* __restrict__ By,
                             const float* __restrict__ emb,
                             const float* __restrict__ Wc, const float* __restrict__ bc,
                             const float* __restrict__ Wm, const float* __restrict__ bm,
                             _Float16* __restrict__ xout) {
  int node = blockIdx.x;
  int b = node >> 9;
  int n = node & 511;
  int e = n >> 1, p = n & 1;
  int t = threadIdx.x;                 // 0..63
  __shared__ float inbuf[64];
  if (t < 32) {
    float cx = p ? Bx[b * EE + e] : Ax[b * EE + e];
    float cy = p ? By[b * EE + e] : Ay[b * EE + e];
    float ct = cx * Wc[t] + cy * Wc[32 + t] + bc[t];
    inbuf[t] = ct > 0.f ? ct : 0.f;
  } else {
    int pid = player[b * 2 + p];
    inbuf[t] = emb[pid * 32 + (t - 32)];
  }
  __syncthreads();
  float acc = bm[t];
  #pragma unroll 8
  for (int k = 0; k < 64; ++k) acc += inbuf[k] * Wm[k * 64 + t];
  xout[(size_t)node * 64 + t] = (_Float16)acc;
}

// -------------------------------------------------------------------------
// GEMM with transposed f16 weights and TRANSPOSED output:
//   hT[b][col][n] = sum_k A[b*512+n][k] * Wt[col][k]
// One wave per 16-row x 16-col tile, K steps of 32 via wmma. All operand
// traffic is 16B vector loads; C store is one v8h per lane.
// -------------------------------------------------------------------------
__global__ void gemm16t_kernel(const _Float16* __restrict__ A,
                               const _Float16* __restrict__ Wt,
                               _Float16* __restrict__ hT, int K) {
  int ft = blockIdx.x & 15;
  int mt = blockIdx.x >> 4;
  int l = threadIdx.x;
  int m = l & 15, hf = l >> 4;
  int col = ft * 16 + m;
  const _Float16* arow = A + (size_t)(mt * 16 + m) * K + hf * 8;
  const _Float16* wrow = Wt + (size_t)col * K + hf * 8;
  v8f acc = {};
  for (int k0 = 0; k0 < K; k0 += 32) {
    if (k0 + 32 < K) {
      __builtin_prefetch(arow + k0 + 32, 0, 3);
      __builtin_prefetch(wrow + k0 + 32, 0, 3);
    }
    v8h alo = *(const v8h*)(arow + k0);
    v8h ahi = *(const v8h*)(arow + k0 + 16);
    v16h a = __builtin_shufflevector(alo, ahi, 0,1,2,3,4,5,6,7,8,9,10,11,12,13,14,15);
    v8h blo = *(const v8h*)(wrow + k0);
    v8h bhi = *(const v8h*)(wrow + k0 + 16);
    v16h bm = __builtin_shufflevector(blo, bhi, 0,1,2,3,4,5,6,7,8,9,10,11,12,13,14,15);
    acc = __builtin_amdgcn_wmma_f32_16x16x32_f16(false, a, false, bm,
                                                 (short)0, acc, false, false);
  }
  int row0 = mt * 16 + hf * 8;         // 8 consecutive node rows
  int b = row0 >> 9;
  int n0 = row0 & 511;
  union { v8h v; _Float16 h[8]; } cU;
  #pragma unroll
  for (int r = 0; r < 8; ++r) cU.h[r] = (_Float16)acc[r];
  *(v8h*)(hT + ((size_t)b * HOUT + col) * NN + n0) = cU.v;
}

// -------------------------------------------------------------------------
// attention scores from transposed h:
//   s_src[b,h,n] = sum_f hT[b][h*64+f][n] * att[h][f];  s_dst with att[h][64+f]
// -------------------------------------------------------------------------
__global__ void score_kernel(const _Float16* __restrict__ hT,
                             const float* __restrict__ att,
                             float* __restrict__ ssrc, float* __restrict__ sdst) {
  int idx = blockIdx.x * blockDim.x + threadIdx.x;   // b*2048 + hh*512 + n
  int n = idx & 511;
  int hh = (idx >> 9) & 3;
  int b = idx >> 11;
  const _Float16* hc = hT + ((size_t)b * HOUT + hh * HID) * NN + n;
  const float* a = att + hh * 2 * HID;
  float ss = 0.f, sd = 0.f;
  #pragma unroll 8
  for (int f = 0; f < HID; ++f) {
    float v = (float)hc[(size_t)f * NN];
    ss += v * a[f];
    sd += v * a[HID + f];
  }
  ssrc[idx] = ss;
  sdst[idx] = sd;
}

// -------------------------------------------------------------------------
// fused masked-softmax attention + P@V via WMMA, online softmax.
// One wave32 per (b, head, 16 query rows). mask == (j != i) (union of all
// 14 relations covers every off-diagonal pair). Scores are rank-1:
// leaky(s_src[i]+s_dst[j]). B operand streams contiguous panels from hT.
// LAYER1: relu + f16 row-major out (feeds next GEMM A); else f32 to d_out.
// -------------------------------------------------------------------------
template <bool LAYER1>
__global__ void attn_kernel(const float* __restrict__ s_src,
                            const float* __restrict__ s_dst,
                            const _Float16* __restrict__ hT,
                            _Float16* __restrict__ outh,
                            float* __restrict__ outf) {
  int it = blockIdx.x & 31;
  int hh = (blockIdx.x >> 5) & 3;
  int b  = blockIdx.x >> 7;
  int l = threadIdx.x;
  int m = l & 15, hf = l >> 4;
  int i0 = it * 16;
  __shared__ float sdst_sh[NN];
  const float* sb = s_dst + ((size_t)b * NHEAD + hh) * NN;
  for (int k = l; k < NN; k += 32) sdst_sh[k] = sb[k];
  float my_ssrc = s_src[((size_t)b * NHEAD + hh) * NN + i0 + m];
  int myrow = i0 + m;
  __syncthreads();

  // per-lane B-panel base pointers: row (col) = hh*64 + ft*16 + m of hT
  const _Float16* hc0 = hT + ((size_t)b * HOUT + hh * HID + m) * NN;
  const _Float16* hc1 = hc0 + (size_t)16 * NN;
  const _Float16* hc2 = hc0 + (size_t)32 * NN;
  const _Float16* hc3 = hc0 + (size_t)48 * NN;

  float run_max = -__builtin_inff();
  float run_sum = 0.f;
  v8f acc0 = {}, acc1 = {}, acc2 = {}, acc3 = {};

  for (int j0 = 0; j0 < NN; j0 += 32) {
    int kb = j0 + hf * 16;
    if (j0 + 32 < NN) {                          // prefetch next B panels
      __builtin_prefetch(hc0 + kb + 32, 0, 3);
      __builtin_prefetch(hc1 + kb + 32, 0, 3);
      __builtin_prefetch(hc2 + kb + 32, 0, 3);
      __builtin_prefetch(hc3 + kb + 32, 0, 3);
    }
    float p[16];
    float tmax = -__builtin_inff();
    #pragma unroll
    for (int e2 = 0; e2 < 16; ++e2) {
      int k = hf * 16 + e2;                      // A-operand K mapping
      int j = j0 + k;
      float sc = my_ssrc + sdst_sh[j];
      sc = sc > 0.f ? sc : 0.2f * sc;            // LeakyReLU(0.2)
      if (j == myrow) sc = -__builtin_inff();    // mask diagonal
      p[e2] = sc;
      tmax = fmaxf(tmax, sc);
    }
    tmax = fmaxf(tmax, __shfl_xor(tmax, 16, 32));
    float nmax = fmaxf(run_max, tmax);
    float scale = __expf(run_max - nmax);
    float psum = 0.f;
    union { v16h v; _Float16 h[16]; } aU;
    #pragma unroll
    for (int e2 = 0; e2 < 16; ++e2) {
      float pe = __expf(p[e2] - nmax);
      psum += pe;
      aU.h[e2] = (_Float16)pe;
    }
    psum += __shfl_xor(psum, 16, 32);
    run_sum = run_sum * scale + psum;
    run_max = nmax;
    // rescale accumulators: C-layout row of VGPR r is hf*8+r; softmax state
    // for that row lives in lane hf*8+r (<16) -> broadcast with shfl.
    #pragma unroll
    for (int r = 0; r < 8; ++r) {
      float rs = __shfl(scale, hf * 8 + r, 32);
      acc0[r] *= rs; acc1[r] *= rs; acc2[r] *= rs; acc3[r] *= rs;
    }
    // B operand: K = kb+e contiguous in hT rows -> two 16B loads per panel
    v8h b0l = *(const v8h*)(hc0 + kb), b0h = *(const v8h*)(hc0 + kb + 8);
    v8h b1l = *(const v8h*)(hc1 + kb), b1h = *(const v8h*)(hc1 + kb + 8);
    v8h b2l = *(const v8h*)(hc2 + kb), b2h = *(const v8h*)(hc2 + kb + 8);
    v8h b3l = *(const v8h*)(hc3 + kb), b3h = *(const v8h*)(hc3 + kb + 8);
    v16h b0 = __builtin_shufflevector(b0l, b0h, 0,1,2,3,4,5,6,7,8,9,10,11,12,13,14,15);
    v16h b1 = __builtin_shufflevector(b1l, b1h, 0,1,2,3,4,5,6,7,8,9,10,11,12,13,14,15);
    v16h b2 = __builtin_shufflevector(b2l, b2h, 0,1,2,3,4,5,6,7,8,9,10,11,12,13,14,15);
    v16h b3 = __builtin_shufflevector(b3l, b3h, 0,1,2,3,4,5,6,7,8,9,10,11,12,13,14,15);
    acc0 = __builtin_amdgcn_wmma_f32_16x16x32_f16(false, aU.v, false, b0, (short)0, acc0, false, false);
    acc1 = __builtin_amdgcn_wmma_f32_16x16x32_f16(false, aU.v, false, b1, (short)0, acc1, false, false);
    acc2 = __builtin_amdgcn_wmma_f32_16x16x32_f16(false, aU.v, false, b2, (short)0, acc2, false, false);
    acc3 = __builtin_amdgcn_wmma_f32_16x16x32_f16(false, aU.v, false, b3, (short)0, acc3, false, false);
  }

  #pragma unroll
  for (int r = 0; r < 8; ++r) {
    float rsum = __shfl(run_sum, hf * 8 + r, 32);
    float inv = 1.f / rsum;
    int row_g = b * NN + i0 + hf * 8 + r;
    int colb = hh * HID + m;
    float v0 = acc0[r] * inv, v1 = acc1[r] * inv, v2 = acc2[r] * inv, v3 = acc3[r] * inv;
    if (LAYER1) {
      v0 = fmaxf(v0, 0.f); v1 = fmaxf(v1, 0.f); v2 = fmaxf(v2, 0.f); v3 = fmaxf(v3, 0.f);
      _Float16* o = outh + (size_t)row_g * HOUT + colb;
      o[0] = (_Float16)v0; o[16] = (_Float16)v1; o[32] = (_Float16)v2; o[48] = (_Float16)v3;
    } else {
      float* o = outf + (size_t)row_g * HOUT + colb;
      o[0] = v0; o[16] = v1; o[32] = v2; o[48] = v3;
    }
  }
}

// -------------------------------------------------------------------------
extern "C" void kernel_launch(void* const* d_in, const int* in_sizes, int n_in,
                              void* d_out, int out_size, void* d_ws, size_t ws_size,
                              hipStream_t stream) {
  const int*   player = (const int*)d_in[0];
  const int*   shot   = (const int*)d_in[1];
  const float* Ax     = (const float*)d_in[2];
  const float* Ay     = (const float*)d_in[3];
  const float* Bx     = (const float*)d_in[4];
  const float* By     = (const float*)d_in[5];
  // d_in[6] = encode_length (hard-coded 256)
  const float* emb    = (const float*)d_in[7];
  const float* Wc     = (const float*)d_in[8];
  const float* bc     = (const float*)d_in[9];
  const float* Wm     = (const float*)d_in[10];
  const float* bm     = (const float*)d_in[11];
  const float* lin1   = (const float*)d_in[12];
  const float* att1   = (const float*)d_in[13];
  const float* lin2   = (const float*)d_in[14];
  const float* att2   = (const float*)d_in[15];

  float* out_emb = (float*)d_out;                       // [B,N,256] f32
  float* out_adj = out_emb + (size_t)BB * NN * HOUT;    // [B,14,N,N] as float

  char* w = (char*)d_ws;
  _Float16* x1f16 = (_Float16*)(w);                                      // 2 MiB
  _Float16* w1t   = (_Float16*)(w + (size_t)(2u  << 20));                // 32 KiB [256,64]
  _Float16* w2t   = (_Float16*)(w + (size_t)(2u  << 20) + (32u << 10));  // 128 KiB [256,256]
  _Float16* h1t   = (_Float16*)(w + (size_t)(3u  << 20));                // 8 MiB [B,256,512]
  float*    s1s   = (float*)   (w + (size_t)(11u << 20));                // 256 KiB
  float*    s1d   = (float*)   (w + (size_t)(11u << 20) + (256u << 10));
  _Float16* x2f16 = (_Float16*)(w + (size_t)(12u << 20));                // 8 MiB [B*N,256]
  _Float16* h2t   = (_Float16*)(w + (size_t)(20u << 20));                // 8 MiB
  float*    s2s   = (float*)   (w + (size_t)(28u << 20));
  float*    s2d   = (float*)   (w + (size_t)(28u << 20) + (256u << 10));

  // 1) adjacency output (dominant HBM traffic: ~470 MB of stores)
  {
    size_t total = (size_t)BB * 14 * NN * (NN / 4);
    adj_kernel<<<(unsigned)(total / 256), 256, 0, stream>>>(shot, out_adj);
  }
  // 2) weight transposes (f32 -> f16, col-major for WMMA B operand)
  wtrans_kernel<<<(256 * 64 + 255) / 256, 256, 0, stream>>>(lin1, w1t, 64);
  wtrans_kernel<<<(256 * 256 + 255) / 256, 256, 0, stream>>>(lin2, w2t, 256);
  // 3) node embedding -> x1 f16
  embed_kernel<<<BB * NN, 64, 0, stream>>>(player, Ax, Ay, Bx, By, emb,
                                           Wc, bc, Wm, bm, x1f16);
  // 4) h1T = (x1 @ lin1_W)^T per batch (K=64)
  gemm16t_kernel<<<(BB * NN / 16) * 16, 32, 0, stream>>>(x1f16, w1t, h1t, 64);
  // 5) per-head attention score terms
  score_kernel<<<BB * NHEAD * NN / 256, 256, 0, stream>>>(h1t, att1, s1s, s1d);
  // 6) GAT layer 1 (relu, f16 row-major out)
  attn_kernel<true><<<BB * NHEAD * (NN / 16), 32, 0, stream>>>(s1s, s1d, h1t, x2f16, nullptr);
  // 7) h2T = (x2 @ lin2_W)^T per batch (K=256)
  gemm16t_kernel<<<(BB * NN / 16) * 16, 32, 0, stream>>>(x2f16, w2t, h2t, 256);
  // 8) scores layer 2
  score_kernel<<<BB * NHEAD * NN / 256, 256, 0, stream>>>(h2t, att2, s2s, s2d);
  // 9) GAT layer 2 -> node_embedding f32 in d_out
  attn_kernel<false><<<BB * NHEAD * (NN / 16), 32, 0, stream>>>(s2s, s2d, h2t, nullptr, out_emb);
  (void)in_sizes; (void)n_in; (void)out_size; (void)ws_size;
}
